// Proj7Exp_34076270526944
// MI455X (gfx1250) — matrix operand order repaired and checked by reading, not compiled, not run on hardware
//
#include <hip/hip_runtime.h>
#include <cstdint>
#include <cstddef>

// ---------------------------------------------------------------------------
// CDNA5 (gfx1250) WMMA types & helpers
// ---------------------------------------------------------------------------
typedef __attribute__((ext_vector_type(16))) _Float16 v16h;
typedef __attribute__((ext_vector_type(8)))  float    v8f;

union AFrag { v16h v; _Float16 h[16]; uint4 q[2]; };
union CFrag { v8f  v; float    f[8];  };

__device__ __forceinline__ v8f wmma_f32_f16(v16h a, v16h b, v8f c) {
  // emits v_wmma_f32_16x16x32_f16
  return __builtin_amdgcn_wmma_f32_16x16x32_f16(false, a, false, b, (short)0, c,
                                                false, false);
}

// ---- async global->LDS copy (CDNA5), with safe fallback -------------------
// Builtin signature (from hipcc diagnostic): param1 is
// 'int __vector_size__(16) __device__ *' i.e. v4i in addrspace(1); LDS side
// is the addrspace(3) analogue.
#if defined(__AMDGCN__) && __has_builtin(__builtin_amdgcn_global_load_async_to_lds_b128) && \
    __has_builtin(__builtin_amdgcn_s_wait_asynccnt)
#define USE_ASYNC_LDS 1
typedef __attribute__((__vector_size__(4 * sizeof(int)))) int v4i_t;
typedef __attribute__((address_space(1))) v4i_t Gv4;
typedef __attribute__((address_space(3))) v4i_t Lv4;
#endif

__device__ __forceinline__ void copy_b128_to_lds(const _Float16* g, _Float16* l) {
#ifdef USE_ASYNC_LDS
  __builtin_amdgcn_global_load_async_to_lds_b128((Gv4*)g, (Lv4*)l, 0, 0);
#else
  *(uint4*)l = *(const uint4*)g;
#endif
}
__device__ __forceinline__ void copy_wait() {
#ifdef USE_ASYNC_LDS
  __builtin_amdgcn_s_wait_asynccnt(0);
#endif
}

__device__ __forceinline__ float gelu_tanh(float x) {  // T5 "gelu_new"
  return 0.5f * x * (1.0f + tanhf(0.7978845608028654f * (x + 0.044715f * x * x * x)));
}
__device__ __forceinline__ float gelu_erf(float x) {   // exact gelu
  return 0.5f * x * (1.0f + erff(x * 0.7071067811865476f));
}

// Problem constants
#define CB 2
#define CC 25
#define CS 256
#define CH 896
#define CNH 12
#define CDK 64
#define CIN 768
#define CDFF 3584
#define CD1 4096
#define CD0 768
#define CN (CB * CC)          // 50 sequences
#define CR (CN * CS)          // 12800 rows
#define CBS (CB * CS)         // 512 rows for MLP3

// ---------------------------------------------------------------------------
// fp32 -> fp16 conversion (weights & activations)
// ---------------------------------------------------------------------------
__global__ void f32_to_f16_kernel(const float* __restrict__ in,
                                  _Float16* __restrict__ out, size_t n) {
  size_t i = (size_t)blockIdx.x * blockDim.x + threadIdx.x;
  if (i < n) out[i] = (_Float16)in[i];
}

// ---------------------------------------------------------------------------
// T5 relative position bias table: bias[NH][S][S]
// ---------------------------------------------------------------------------
__global__ void bias_kernel(const float* __restrict__ rel_emb,
                            float* __restrict__ bias) {
  int idx = blockIdx.x * blockDim.x + threadIdx.x;
  if (idx >= CS * CS) return;
  int i = idx >> 8, j = idx & 255;
  int rel = j - i;
  int ret = (rel > 0) ? 16 : 0;
  int nmag = (rel < 0) ? -rel : rel;
  int bucket;
  if (nmag < 8) {
    bucket = ret + nmag;
  } else {
    int large = 8 + (int)(logf((float)nmag * 0.125f) * (1.0f / logf(16.0f)) * 8.0f);
    if (large > 15) large = 15;
    bucket = ret + large;
  }
  for (int hh = 0; hh < CNH; ++hh)
    bias[((size_t)hh * CS + i) * CS + j] = rel_emb[bucket * CNH + hh];
}

// ---------------------------------------------------------------------------
// RMSNorm (optionally emits f16 and/or f32)
// ---------------------------------------------------------------------------
__global__ __launch_bounds__(256) void rmsnorm_kernel(
    const float* __restrict__ x, const float* __restrict__ w,
    _Float16* outh, float* outf, int H) {
  const int row = blockIdx.x;
  const float* xr = x + (size_t)row * H;
  const int tid = threadIdx.x, wave = tid >> 5, lane = tid & 31;
  float ss = 0.f;
  for (int i = tid; i < H; i += 256) { float t = xr[i]; ss += t * t; }
#pragma unroll
  for (int o = 16; o > 0; o >>= 1) ss += __shfl_xor(ss, o, 32);  // wave32
  __shared__ float red[8];
  if (lane == 0) red[wave] = ss;
  __syncthreads();
  if (tid == 0) { float t = 0.f; for (int i = 0; i < 8; ++i) t += red[i]; red[0] = t; }
  __syncthreads();
  const float scale = rsqrtf(red[0] / (float)H + 1e-6f);
  for (int i = tid; i < H; i += 256) {
    const float vv = xr[i] * scale * w[i];
    if (outh) outh[(size_t)row * H + i] = (_Float16)vv;
    if (outf) outf[(size_t)row * H + i] = vv;
  }
}

// ---------------------------------------------------------------------------
// LayerNorm -> f16
// ---------------------------------------------------------------------------
__global__ __launch_bounds__(256) void layernorm_kernel(
    const float* __restrict__ x, const float* __restrict__ w,
    const float* __restrict__ b, _Float16* __restrict__ out, int H) {
  const int row = blockIdx.x;
  const float* xr = x + (size_t)row * H;
  const int tid = threadIdx.x, wave = tid >> 5, lane = tid & 31;
  float s1 = 0.f, s2 = 0.f;
  for (int i = tid; i < H; i += 256) { float t = xr[i]; s1 += t; s2 += t * t; }
#pragma unroll
  for (int o = 16; o > 0; o >>= 1) {
    s1 += __shfl_xor(s1, o, 32);
    s2 += __shfl_xor(s2, o, 32);
  }
  __shared__ float r1[8], r2[8];
  if (lane == 0) { r1[wave] = s1; r2[wave] = s2; }
  __syncthreads();
  if (tid == 0) {
    float a = 0.f, c = 0.f;
    for (int i = 0; i < 8; ++i) { a += r1[i]; c += r2[i]; }
    r1[0] = a; r2[0] = c;
  }
  __syncthreads();
  const float mu = r1[0] / (float)H;
  const float var = r2[0] / (float)H - mu * mu;
  const float inv = rsqrtf(var + 1e-6f);
  for (int i = tid; i < H; i += 256)
    out[(size_t)row * H + i] = (_Float16)((xr[i] - mu) * inv * w[i] + b[i]);
}

// ---------------------------------------------------------------------------
// Channel-scale mean over C:  h2[b,s,h] = mean_c(cha_scale[c] * hn[b,c,s,h])
// ---------------------------------------------------------------------------
__global__ void chan_mean_kernel(const float* __restrict__ hn,
                                 const float* __restrict__ cs,
                                 float* __restrict__ h2) {
  int idx = blockIdx.x * blockDim.x + threadIdx.x;
  if (idx >= CBS * CH) return;
  int hh = idx % CH;
  int s = (idx / CH) % CS;
  int b = idx / (CH * CS);
  float sum = 0.f;
  for (int c = 0; c < CC; ++c)
    sum += cs[c] * hn[(((size_t)(b * CC + c) * CS + s) * CH) + hh];
  h2[idx] = sum * (1.0f / (float)CC);
}

// exact-GELU elementwise f32 -> f16
__global__ void gelu_exact_kernel(const float* __restrict__ in,
                                  _Float16* __restrict__ out, size_t n) {
  size_t i = (size_t)blockIdx.x * blockDim.x + threadIdx.x;
  if (i < n) out[i] = (_Float16)gelu_erf(in[i]);
}

// mean over S: x1[b,j] = mean_s y[b*S+s, j]
__global__ void mean_s_kernel(const float* __restrict__ y, float* __restrict__ out) {
  int idx = blockIdx.x * blockDim.x + threadIdx.x;
  if (idx >= CB * CD0) return;
  int b = idx / CD0, j = idx % CD0;
  float s = 0.f;
  for (int t = 0; t < CS; ++t) s += y[(size_t)(b * CS + t) * CD0 + j];
  out[idx] = s * (1.0f / (float)CS);
}

// ---------------------------------------------------------------------------
// Generic WMMA GEMM (MODE is compile-time; no runtime branches in hot loop).
// Block tile 128x64, 8 waves (wave32), each wave 2x2 16x16 tiles, K-step 32.
// A tile staged row-major (async global->LDS); B tile staged TRANSPOSED in
// LDS so every fragment is two ds_load_b128's:
//   A 16x32 frag: runs [hi*8 .. +7] and [16+hi*8 .. +7] of the A row.
//   B 32x16 frag: runs [hi*8 .. +7] and [16+hi*8 .. +7] of the transposed B column row.
// Modes:
//  0: outH = A@B0
//  1: outF = resid + A@B0           (residual add, fp32)
//  2: outH = gelu_exact(A@B0)
//  3: outH = gelu_tanh(A@B0) * (A@B1)   (gated FFN, dual-B)
//  4: outF = A@B0 + bvec[col]
//  5: outF = A@B0
// Requires M % 128 == 0, N % 64 == 0, K % 32 == 0.
// ---------------------------------------------------------------------------
#define GBM 128
#define GBN 64
#define GBK 32

template <int MODE>
__global__ __launch_bounds__(256) void gemm_wmma_kernel(
    const _Float16* __restrict__ A, const _Float16* __restrict__ B0,
    const _Float16* __restrict__ B1, float* outF, _Float16* outH,
    const float* resid, const float* __restrict__ bvec, int M, int N, int K) {
  __shared__ __align__(16) _Float16 As[GBM][GBK];
  __shared__ __align__(16) _Float16 Bst0[GBN][GBK];  // transposed
  __shared__ __align__(16) _Float16 Bst1[GBN][GBK];  // transposed (gated)

  const int tid = threadIdx.x;
  const int wave = tid >> 5, lane = tid & 31;
  const int mlane = lane & 15, hi = lane >> 4;
  const int bm = blockIdx.x * GBM, bn = blockIdx.y * GBN;
  const int wm = (wave & 3) * 32, wn = (wave >> 2) * 32;

  CFrag acc0[2][2], acc1[2][2];
#pragma unroll
  for (int i = 0; i < 2; ++i)
#pragma unroll
    for (int j = 0; j < 2; ++j)
#pragma unroll
      for (int e = 0; e < 8; ++e) { acc0[i][j].f[e] = 0.f; acc1[i][j].f[e] = 0.f; }

  const int ar = tid >> 1, ac = (tid & 1) * 16;   // A tile: 128 x 32
  const int br = tid >> 3, bc = (tid & 7) * 8;    // B tile:  32 x 64

  for (int k0 = 0; k0 < K; k0 += GBK) {
    // A: async b128 global -> LDS, row-major
    const _Float16* aS = A + (size_t)(bm + ar) * K + k0 + ac;
    copy_b128_to_lds(aS, &As[ar][ac]);
    copy_b128_to_lds(aS + 8, &As[ar][ac + 8]);
    // B: coalesced b128 load + register transpose into LDS
    {
      uint4 tb = *(const uint4*)(B0 + (size_t)(k0 + br) * N + bn + bc);
      const _Float16* tbh = (const _Float16*)&tb;
#pragma unroll
      for (int j = 0; j < 8; ++j) Bst0[bc + j][br] = tbh[j];
      if (MODE == 3) {
        uint4 tb1 = *(const uint4*)(B1 + (size_t)(k0 + br) * N + bn + bc);
        const _Float16* tbh1 = (const _Float16*)&tb1;
#pragma unroll
        for (int j = 0; j < 8; ++j) Bst1[bc + j][br] = tbh1[j];
      }
    }
    if (k0 + GBK < K) {  // global_prefetch_b8 for next K tile
      __builtin_prefetch(A + (size_t)(bm + ar) * K + k0 + GBK + ac, 0, 0);
      __builtin_prefetch(B0 + (size_t)(k0 + GBK + br) * N + bn + bc, 0, 0);
    }
    copy_wait();
    __syncthreads();

    AFrag af[2], bf0[2], bf1[2];
#pragma unroll
    for (int t = 0; t < 2; ++t) {
      const _Float16* arow = &As[wm + t * 16 + mlane][0];
      af[t].q[0] = *(const uint4*)(arow + hi * 8);
      af[t].q[1] = *(const uint4*)(arow + 16 + hi * 8);
      const _Float16* brow = &Bst0[wn + t * 16 + mlane][0];
      bf0[t].q[0] = *(const uint4*)(brow + hi * 8);
      bf0[t].q[1] = *(const uint4*)(brow + 16 + hi * 8);
      if (MODE == 3) {
        const _Float16* brow1 = &Bst1[wn + t * 16 + mlane][0];
        bf1[t].q[0] = *(const uint4*)(brow1 + hi * 8);
        bf1[t].q[1] = *(const uint4*)(brow1 + 16 + hi * 8);
      }
    }
#pragma unroll
    for (int tm = 0; tm < 2; ++tm)
#pragma unroll
      for (int tn = 0; tn < 2; ++tn) {
        acc0[tm][tn].v = wmma_f32_f16(af[tm].v, bf0[tn].v, acc0[tm][tn].v);
        if (MODE == 3)
          acc1[tm][tn].v = wmma_f32_f16(af[tm].v, bf1[tn].v, acc1[tm][tn].v);
      }
    __syncthreads();
  }

#pragma unroll
  for (int tm = 0; tm < 2; ++tm)
    for (int tn = 0; tn < 2; ++tn)
#pragma unroll
      for (int j = 0; j < 8; ++j) {
        const int row = bm + wm + tm * 16 + hi * 8 + j;   // C/D layout
        const int col = bn + wn + tn * 16 + mlane;
        const size_t off = (size_t)row * N + col;
        const float a = acc0[tm][tn].f[j];
        if (MODE == 0) outH[off] = (_Float16)a;
        else if (MODE == 1) outF[off] = resid[off] + a;
        else if (MODE == 2) outH[off] = (_Float16)gelu_erf(a);
        else if (MODE == 3) outH[off] = (_Float16)(gelu_tanh(a) * acc1[tm][tn].f[j]);
        else if (MODE == 4) outF[off] = a + bvec[col];
        else outF[off] = a;
      }
}

// ---------------------------------------------------------------------------
// Attention: one block per (sequence n, head h).  4 waves / 128 threads.
// LDS (160KB dyn): K[256][64] (32K, async-staged) + Vt[64][256] (32K,
// transposed) + fp32 scores 4x16x256 (64K) + f16 probs 4x16x256 (32K).
// scores = Q @ K^T + bias, wave32 shuffle softmax, ctx = P @ V.
// All WMMA fragments are 2 x b128 LDS (or global) loads.
// ---------------------------------------------------------------------------
__global__ __launch_bounds__(128) void attn_wmma_kernel(
    const _Float16* __restrict__ q, const _Float16* __restrict__ k,
    const _Float16* __restrict__ v, const float* __restrict__ bias,
    _Float16* __restrict__ ctx) {
  extern __shared__ __align__(16) char smem[];
  _Float16(*Ksh)[64] = (_Float16(*)[64])smem;                      // [key][d]
  _Float16(*Vt)[CS] = (_Float16(*)[CS])(smem + 32 * 1024);         // [d][key]
  float* Ssc = (float*)(smem + 64 * 1024);                         // 4*16*256 f32
  _Float16* Pb = (_Float16*)(smem + 128 * 1024);                   // 4*16*256 f16

  const int n = blockIdx.x / CNH, h = blockIdx.x % CNH;
  const int tid = threadIdx.x, wave = tid >> 5, lane = tid & 31;
  const int mlane = lane & 15, hi = lane >> 4;
  const size_t rowbase = (size_t)n * CS * CIN + (size_t)h * CDK;

  for (int r = tid; r < CS; r += 128) {
    const _Float16* ksrc = k + rowbase + (size_t)r * CIN;
    const _Float16* vsrc = v + rowbase + (size_t)r * CIN;
#pragma unroll
    for (int c = 0; c < 64; c += 8) {
      copy_b128_to_lds(ksrc + c, &Ksh[r][c]);  // async, row-major
      uint4 tv = *(const uint4*)(vsrc + c);    // transpose via registers
      const _Float16* tvh = (const _Float16*)&tv;
#pragma unroll
      for (int j = 0; j < 8; ++j) Vt[c + j][r] = tvh[j];
    }
  }
  copy_wait();
  __syncthreads();

  float* Sw = Ssc + (size_t)wave * 16 * CS;
  _Float16* Pw = Pb + (size_t)wave * 16 * CS;

  for (int qt = wave; qt < 16; qt += 4) {
    const int qbase = qt * 16;
    // Q fragments (K-dim 64 -> 2 steps of 32), two b128 global loads each
    AFrag aq[2];
    {
      const _Float16* qrow = q + rowbase + (size_t)(qbase + mlane) * CIN;
#pragma unroll
      for (int ks = 0; ks < 2; ++ks) {
        aq[ks].q[0] = *(const uint4*)(qrow + ks * 32 + hi * 8);
        aq[ks].q[1] = *(const uint4*)(qrow + ks * 32 + 16 + hi * 8);
      }
    }
    // scores = Q @ K^T + bias  -> LDS (fp32)
    for (int nt = 0; nt < 16; ++nt) {
      CFrag acc;
#pragma unroll
      for (int e = 0; e < 8; ++e) acc.f[e] = 0.f;
      const _Float16* krow = &Ksh[nt * 16 + mlane][0];
#pragma unroll
      for (int ks = 0; ks < 2; ++ks) {
        AFrag bf;
        bf.q[0] = *(const uint4*)(krow + ks * 32 + hi * 8);
        bf.q[1] = *(const uint4*)(krow + ks * 32 + 16 + hi * 8);
        acc.v = wmma_f32_f16(aq[ks].v, bf.v, acc.v);
      }
#pragma unroll
      for (int j = 0; j < 8; ++j) {
        const int lr = hi * 8 + j;
        Sw[lr * CS + nt * 16 + mlane] =
            acc.f[j] +
            bias[((size_t)h * CS + (qbase + lr)) * CS + nt * 16 + mlane];
      }
    }
    // softmax (wave32 shuffle reduce); normalized probs written as f16
    for (int r = 0; r < 16; ++r) {
      float* row = Sw + r * CS;
      float mx = -3.0e38f;
      for (int c = lane; c < CS; c += 32) mx = fmaxf(mx, row[c]);
#pragma unroll
      for (int o = 16; o > 0; o >>= 1) mx = fmaxf(mx, __shfl_xor(mx, o, 32));
      float sum = 0.f;
      for (int c = lane; c < CS; c += 32) {
        float e = __expf(row[c] - mx);
        row[c] = e;
        sum += e;
      }
#pragma unroll
      for (int o = 16; o > 0; o >>= 1) sum += __shfl_xor(sum, o, 32);
      const float inv = 1.0f / sum;
      for (int c = lane; c < CS; c += 32)
        Pw[r * CS + c] = (_Float16)(row[c] * inv);
    }
    // ctx = P @ V  (K-dim 256 -> 8 steps; d 64 -> 4 n-tiles)
    for (int nt = 0; nt < 4; ++nt) {
      CFrag acc;
#pragma unroll
      for (int e = 0; e < 8; ++e) acc.f[e] = 0.f;
      const _Float16* prow = Pw + mlane * CS;
      const _Float16* vrow = &Vt[nt * 16 + mlane][0];
#pragma unroll
      for (int ks = 0; ks < 8; ++ks) {
        AFrag af, bf;
        af.q[0] = *(const uint4*)(prow + ks * 32 + hi * 8);
        af.q[1] = *(const uint4*)(prow + ks * 32 + 16 + hi * 8);
        bf.q[0] = *(const uint4*)(vrow + ks * 32 + hi * 8);
        bf.q[1] = *(const uint4*)(vrow + ks * 32 + 16 + hi * 8);
        acc.v = wmma_f32_f16(af.v, bf.v, acc.v);
      }
#pragma unroll
      for (int j = 0; j < 8; ++j) {
        const int qq = qbase + hi * 8 + j;
        ctx[rowbase + (size_t)qq * CIN + nt * 16 + mlane] = (_Float16)acc.f[j];
      }
    }
  }
}

// ---------------------------------------------------------------------------
// Orchestration
// ---------------------------------------------------------------------------
extern "C" void kernel_launch(void* const* d_in, const int* in_sizes, int n_in,
                              void* d_out, int out_size, void* d_ws,
                              size_t ws_size, hipStream_t stream) {
  (void)in_sizes; (void)n_in; (void)out_size; (void)ws_size;

  const float* x_in      = (const float*)d_in[0];
  const float* cha_scale = (const float*)d_in[1];
  const float* rel_emb   = (const float*)d_in[2];
  const float* L[2][9];
  for (int l = 0; l < 2; ++l)
    for (int j = 0; j < 9; ++j) L[l][j] = (const float*)d_in[3 + l * 9 + j];
  const float* final_ln = (const float*)d_in[21];
  const float* mlp_ln_w = (const float*)d_in[22];
  const float* mlp_ln_b = (const float*)d_in[23];
  const float* p1 = (const float*)d_in[24];
  const float* p2 = (const float*)d_in[25];
  const float* fc_w = (const float*)d_in[26];
  const float* fc_b = (const float*)d_in[27];

  // workspace carve-up (256B aligned)
  char* p = (char*)d_ws;
  auto alloc = [&](size_t bytes) -> void* {
    void* r = (void*)p;
    p += (bytes + 255) & ~(size_t)255;
    return r;
  };
  _Float16 *wqh[2], *wkh[2], *wvh[2], *woh[2], *wi0h[2], *wi1h[2], *wouth[2];
  for (int l = 0; l < 2; ++l) {
    wqh[l]   = (_Float16*)alloc((size_t)CH * CIN * 2);
    wkh[l]   = (_Float16*)alloc((size_t)CH * CIN * 2);
    wvh[l]   = (_Float16*)alloc((size_t)CH * CIN * 2);
    woh[l]   = (_Float16*)alloc((size_t)CIN * CH * 2);
    wi0h[l]  = (_Float16*)alloc((size_t)CH * CDFF * 2);
    wi1h[l]  = (_Float16*)alloc((size_t)CH * CDFF * 2);
    wouth[l] = (_Float16*)alloc((size_t)CDFF * CH * 2);
  }
  _Float16* p1h = (_Float16*)alloc((size_t)CH * CD1 * 2);
  _Float16* p2h = (_Float16*)alloc((size_t)CD1 * CD1 * 2);
  _Float16* fch = (_Float16*)alloc((size_t)CD1 * CD0 * 2);
  float* biasb  = (float*)alloc((size_t)CNH * CS * CS * 4);
  float* xbuf   = (float*)alloc((size_t)CR * CH * 4);
  _Float16* hnh = (_Float16*)alloc((size_t)CR * CH * 2);
  _Float16* qh  = (_Float16*)alloc((size_t)CR * CIN * 2);
  _Float16* kh  = (_Float16*)alloc((size_t)CR * CIN * 2);
  _Float16* vh  = (_Float16*)alloc((size_t)CR * CIN * 2);
  _Float16* ctxh = (_Float16*)alloc((size_t)CR * CIN * 2);
  _Float16* ffh  = (_Float16*)alloc((size_t)CR * CDFF * 2);
  float* hnf = (float*)ffh;  // reuse FFN buffer for final RMSNorm fp32 output
  float* h2      = (float*)alloc((size_t)CBS * CH * 4);
  _Float16* hn2h = (_Float16*)alloc((size_t)CBS * CH * 2);
  _Float16* m1h  = (_Float16*)alloc((size_t)CBS * CD1 * 2);
  _Float16* gx2h = (_Float16*)alloc((size_t)CBS * CD1 * 2);
  float* ytmp    = (float*)alloc((size_t)CBS * CD0 * 4);

  float* x1out = (float*)d_out;                 // [2, 768]
  float* x2out = (float*)d_out + CB * CD0;      // [2, 256, 4096]

  auto cvt = [&](const float* src, _Float16* dst, size_t n) {
    f32_to_f16_kernel<<<dim3((unsigned)((n + 255) / 256)), 256, 0, stream>>>(src, dst, n);
  };
  // weights -> f16
  for (int l = 0; l < 2; ++l) {
    cvt(L[l][1], wqh[l],   (size_t)CH * CIN);
    cvt(L[l][2], wkh[l],   (size_t)CH * CIN);
    cvt(L[l][3], wvh[l],   (size_t)CH * CIN);
    cvt(L[l][4], woh[l],   (size_t)CIN * CH);
    cvt(L[l][6], wi0h[l],  (size_t)CH * CDFF);
    cvt(L[l][7], wi1h[l],  (size_t)CH * CDFF);
    cvt(L[l][8], wouth[l], (size_t)CDFF * CH);
  }
  cvt(p1,   p1h, (size_t)CH * CD1);
  cvt(p2,   p2h, (size_t)CD1 * CD1);
  cvt(fc_w, fch, (size_t)CD1 * CD0);

  bias_kernel<<<dim3((CS * CS + 255) / 256), 256, 0, stream>>>(rel_emb, biasb);
  (void)hipMemcpyAsync(xbuf, x_in, (size_t)CR * CH * 4, hipMemcpyDeviceToDevice,
                       stream);

  const unsigned attnLds = 160 * 1024;  // 32K K + 32K Vt + 64K scores + 32K probs

  for (int l = 0; l < 2; ++l) {
    // --- self-attention ---
    rmsnorm_kernel<<<CR, 256, 0, stream>>>(xbuf, L[l][0], hnh, nullptr, CH);
    gemm_wmma_kernel<0><<<dim3(CR / GBM, CIN / GBN), 256, 0, stream>>>(
        hnh, wqh[l], nullptr, nullptr, qh, nullptr, nullptr, CR, CIN, CH);
    gemm_wmma_kernel<0><<<dim3(CR / GBM, CIN / GBN), 256, 0, stream>>>(
        hnh, wkh[l], nullptr, nullptr, kh, nullptr, nullptr, CR, CIN, CH);
    gemm_wmma_kernel<0><<<dim3(CR / GBM, CIN / GBN), 256, 0, stream>>>(
        hnh, wvh[l], nullptr, nullptr, vh, nullptr, nullptr, CR, CIN, CH);
    attn_wmma_kernel<<<dim3(CN * CNH), 128, attnLds, stream>>>(qh, kh, vh, biasb, ctxh);
    gemm_wmma_kernel<1><<<dim3(CR / GBM, CH / GBN), 256, 0, stream>>>(
        ctxh, woh[l], nullptr, xbuf, nullptr, xbuf, nullptr, CR, CH, CIN);
    // --- gated-gelu FFN ---
    rmsnorm_kernel<<<CR, 256, 0, stream>>>(xbuf, L[l][5], hnh, nullptr, CH);
    gemm_wmma_kernel<3><<<dim3(CR / GBM, CDFF / GBN), 256, 0, stream>>>(
        hnh, wi0h[l], wi1h[l], nullptr, ffh, nullptr, nullptr, CR, CDFF, CH);
    gemm_wmma_kernel<1><<<dim3(CR / GBM, CH / GBN), 256, 0, stream>>>(
        ffh, wouth[l], nullptr, xbuf, nullptr, xbuf, nullptr, CR, CH, CDFF);
  }

  // final RMSNorm -> fp32 (reuses FFN buffer), channel-scale mean, LayerNorm
  rmsnorm_kernel<<<CR, 256, 0, stream>>>(xbuf, final_ln, nullptr, hnf, CH);
  chan_mean_kernel<<<dim3((CBS * CH + 255) / 256), 256, 0, stream>>>(hnf, cha_scale, h2);
  layernorm_kernel<<<CBS, 256, 0, stream>>>(h2, mlp_ln_w, mlp_ln_b, hn2h, CH);

  // MLP3 head
  gemm_wmma_kernel<2><<<dim3(CBS / GBM, CD1 / GBN), 256, 0, stream>>>(
      hn2h, p1h, nullptr, nullptr, m1h, nullptr, nullptr, CBS, CD1, CH);
  gemm_wmma_kernel<5><<<dim3(CBS / GBM, CD1 / GBN), 256, 0, stream>>>(
      m1h, p2h, nullptr, x2out, nullptr, nullptr, nullptr, CBS, CD1, CD1);
  gelu_exact_kernel<<<dim3((CBS * CD1 + 255) / 256), 256, 0, stream>>>(
      x2out, gx2h, (size_t)CBS * CD1);
  gemm_wmma_kernel<4><<<dim3(CBS / GBM, CD0 / GBN), 256, 0, stream>>>(
      gx2h, fch, nullptr, ytmp, nullptr, nullptr, fc_b, CBS, CD0, CD1);
  mean_s_kernel<<<dim3((CB * CD0 + 255) / 256), 256, 0, stream>>>(ytmp, x1out);
}
